// StructuredPredictionLayer_13391708028985
// MI455X (gfx1250) — matrix operand order repaired and checked by reading, not compiled, not run on hardware
//
#include <hip/hip_runtime.h>
#include <hip/hip_bf16.h>

typedef __attribute__((ext_vector_type(16))) __bf16 v16bf;
typedef __attribute__((ext_vector_type(8)))  float  v8f;
typedef __attribute__((ext_vector_type(4)))  int    v4i;

#define AS_GLOBAL __attribute__((address_space(1)))
#define AS_LDS    __attribute__((address_space(3)))

#define FDIM    512
#define HID     128
#define NJ      24
#define JD      6
#define IN_DIM  518      // 512 + 6
#define NKT     17       // ceil(544/32) k-tiles of 32
#define ROWS    128      // batch rows per workgroup
#define THREADS 256      // 8 wave32

// LDS strides (elements)
#define SF_STRIDE  552   // 544 + 8 pad (bf16)
#define SW_STRIDE  40    // 32 + 8     (bf16, transposed W1 tile [HID][40])
#define SW2_STRIDE 136   // 128 + 8    (bf16, transposed W2 [16][136])
#define SH_STRIDE  136   // 128 + 8    (bf16, H scratch [ROWS][136])
#define SP_STRIDE  148   // 144 + 4    (f32, predictions [ROWS][148])

#define SF_BYTES  (ROWS * SF_STRIDE * 2)            // 141312
#define SW_BYTES  (2 * HID * SW_STRIDE * 2)         // 20480
#define SW2_BYTES (16 * SW2_STRIDE * 2)             // 4352
#define SH_BYTES  (ROWS * SH_STRIDE * 2)            // 34816
#define SP_BYTES  (ROWS * SP_STRIDE * 4)            // 75776
#define LDS_BYTES (SF_BYTES + SW_BYTES + SW2_BYTES + SH_BYTES + SP_BYTES) // 276736

// Workspace layout: tile-contiguous bf16 weights
#define WT1_TILE_ELEMS (HID * 32)                    // 4096 bf16 = 8 KB per k-tile
#define WT1_ELEMS      ((size_t)NJ * NKT * WT1_TILE_ELEMS)
#define WT2_JOINT_ELEMS (16 * HID)                   // 2048 bf16 = 4 KB per joint
#define WT2_ELEMS      ((size_t)NJ * WT2_JOINT_ELEMS)

#if defined(__has_builtin)
#if __has_builtin(__builtin_amdgcn_global_load_async_to_lds_b128) && \
    __has_builtin(__builtin_amdgcn_s_wait_asynccnt)
#define HAS_ASYNC_LDS 1
#endif
#endif
#ifndef HAS_ASYNC_LDS
#define HAS_ASYNC_LDS 0
#endif

__constant__ int c_parents[NJ] = {-1, 0, 0, 0, 1, 2, 3, 4, 5, 6, 7, 8,
                                   9, 9, 9, 12, 13, 14, 16, 17, 18, 19, 20, 21};

// hardware f32 -> bf16 (RNE) via v_cvt, bit pattern out
__device__ __forceinline__ unsigned short bfbits(float f) {
    union { __bf16 b; unsigned short u; } cv;
    cv.b = (__bf16)f;
    return cv.u;
}
__device__ __forceinline__ unsigned int pack_bf2(float lo, float hi) {
    return (unsigned int)bfbits(lo) | ((unsigned int)bfbits(hi) << 16);
}

union Frag {
    uint4 q[2];
    v16bf v;
};

// A fragment (M=16 x K=32 bf16), row-major LDS source.
// Lane L: row M = L&15; K-half offset 8 for high lanes; two b128 chunks.
__device__ __forceinline__ v16bf load_a(const unsigned short* lds, int row0,
                                        int stride, int kbase, int lane) {
    int r  = row0 + (lane & 15);
    int kh = (lane & 16) ? 8 : 0;
    const unsigned short* p = lds + r * stride + kbase + kh;
    Frag f;
    f.q[0] = *(const uint4*)(p);
    f.q[1] = *(const uint4*)(p + 16);
    return f.v;
}

// B fragment (K=32 x N=16 bf16), TRANSPOSED LDS source ([N][K]).
// Lane L: col N = L&15; K-half offset 8 for high lanes; two b128 chunks (matches load_a).
__device__ __forceinline__ v16bf load_b(const unsigned short* lds, int ncol0,
                                        int stride, int kbase, int lane) {
    int n  = ncol0 + (lane & 15);
    int kh = (lane & 16) ? 8 : 0;
    const unsigned short* p = lds + n * stride + kbase + kh;
    Frag f;
    f.q[0] = *(const uint4*)(p);
    f.q[1] = *(const uint4*)(p + 16);
    return f.v;
}

__device__ __forceinline__ v8f zero8() {
    v8f z;
#pragma unroll
    for (int i = 0; i < 8; ++i) z[i] = 0.0f;
    return z;
}

// 16-byte chunk copy global(bf16, packed) -> LDS(bf16, padded stride),
// async DMA if available (ASYNCcnt), else load+ds_store.
__device__ __forceinline__ void copy_chunk16(const unsigned short* g,
                                             unsigned short* l) {
#if HAS_ASYNC_LDS
    __builtin_amdgcn_global_load_async_to_lds_b128(
        (AS_GLOBAL v4i*)g, (AS_LDS v4i*)l, 0, 0);
#else
    uint4 v = *(const uint4*)g;
    *(uint4*)l = v;
#endif
}

__device__ __forceinline__ void async_barrier() {
#if HAS_ASYNC_LDS
    __builtin_amdgcn_s_wait_asynccnt(0);
#endif
    __syncthreads();
}

// ---------------- prep: convert + transpose weights once ----------------
// W1 [NJ][IN_DIM][HID] f32 -> wt1 [NJ][NKT][HID][32] bf16 (zero-padded K)
extern "C" __global__ __launch_bounds__(THREADS)
void prep_w1_kernel(const float* __restrict__ W1, unsigned short* __restrict__ wt1) {
    const int tile = blockIdx.x;            // j * NKT + kt
    const int j  = tile / NKT;
    const int kt = tile % NKT;
    const float* w = W1 + (size_t)j * IN_DIM * HID;
    const int n  = threadIdx.x & 127;
    const int k0 = (threadIdx.x >> 7) * 16; // 0 or 16
    unsigned int pk[8];
#pragma unroll
    for (int i = 0; i < 16; i += 2) {
        int k = kt * 32 + k0 + i;
        float a = (k     < IN_DIM) ? w[(size_t)k * HID + n]       : 0.0f;
        float b = (k + 1 < IN_DIM) ? w[(size_t)(k + 1) * HID + n] : 0.0f;
        pk[i >> 1] = pack_bf2(a, b);
    }
    unsigned short* d = wt1 + (size_t)tile * WT1_TILE_ELEMS + n * 32 + k0;
    ((uint4*)d)[0] = make_uint4(pk[0], pk[1], pk[2], pk[3]);
    ((uint4*)d)[1] = make_uint4(pk[4], pk[5], pk[6], pk[7]);
}

// W2 [NJ][HID][JD] f32 -> wt2 [NJ][16][HID] bf16 (transposed, N padded to 16)
extern "C" __global__ __launch_bounds__(THREADS)
void prep_w2_kernel(const float* __restrict__ W2, unsigned short* __restrict__ wt2) {
    const int j = blockIdx.x;
    const float* w = W2 + (size_t)j * HID * JD;
    for (int idx = threadIdx.x; idx < WT2_JOINT_ELEMS; idx += THREADS) {
        int n = idx >> 7;        // 0..15
        int k = idx & 127;
        float v = (n < JD) ? w[k * JD + n] : 0.0f;
        wt2[(size_t)j * WT2_JOINT_ELEMS + n * HID + k] = bfbits(v);
    }
}

// ---------------- fused chain kernel ----------------
extern "C" __global__ __launch_bounds__(THREADS)
void smpl_chain_kernel(const float* __restrict__ features,
                       const unsigned short* __restrict__ wt1,
                       const unsigned short* __restrict__ wt2,
                       const float* __restrict__ b1,
                       const float* __restrict__ b2,
                       float* __restrict__ out) {
    extern __shared__ char lds_raw[];
    unsigned short* sF  = (unsigned short*)(lds_raw);
    unsigned short* sWt = (unsigned short*)(lds_raw + SF_BYTES);
    unsigned short* sW2 = (unsigned short*)(lds_raw + SF_BYTES + SW_BYTES);
    unsigned short* sH  = (unsigned short*)(lds_raw + SF_BYTES + SW_BYTES + SW2_BYTES);
    float*          sP  = (float*)(lds_raw + SF_BYTES + SW_BYTES + SW2_BYTES + SH_BYTES);

    const int tid  = threadIdx.x;
    const int lane = tid & 31;
    const int wave = tid >> 5;
    const int n15  = lane & 15;
    const int hi8  = (lane & 16) ? 8 : 0;
    const int row0 = blockIdx.x * ROWS;

    // ---- stage features -> bf16 LDS tile; zero rot/pad columns ----
    {
        const float* fsrc = features + (size_t)row0 * FDIM;
        for (int idx = tid * 4; idx < ROWS * FDIM; idx += THREADS * 4) {
            int r = idx >> 9;
            int c = idx & 511;
            float4 v = *(const float4*)(fsrc + (size_t)r * FDIM + c);
            *(uint2*)(sF + r * SF_STRIDE + c) =
                make_uint2(pack_bf2(v.x, v.y), pack_bf2(v.z, v.w));
        }
        const int padw = SF_STRIDE - FDIM;  // rot tile 512..543 + row pad
        for (int idx = tid; idx < ROWS * padw; idx += THREADS) {
            int r = idx / padw;
            int c = FDIM + idx % padw;
            sF[r * SF_STRIDE + c] = 0;
        }
    }

    // stage one prepped 8KB W1 k-tile into sWt[buf] (pad stride 40)
    auto stage_w1_tile = [&](int j, int kt, int buf) {
        const unsigned short* g = wt1 + ((size_t)j * NKT + kt) * WT1_TILE_ELEMS;
        unsigned short* l = sWt + buf * (HID * SW_STRIDE);
#pragma unroll
        for (int cc = 0; cc < 2; ++cc) {
            int c  = tid + cc * THREADS;    // 512 chunks of 16B
            int n  = c >> 2;
            int k8 = (c & 3) * 8;
            copy_chunk16(g + c * 8, l + n * SW_STRIDE + k8);
        }
    };

    v8f acc[8];

    for (int j = 0; j < NJ; ++j) {
        async_barrier();  // sP / rot cols / sW2 / sWt buf0 from prev joint done

        // parent rotation -> K-tile 16 columns of sF (6 live cols)
        {
            int p = c_parents[j];
            if (p >= 0 && tid < ROWS) {
                const float* src = sP + tid * SP_STRIDE + p * JD;
                unsigned short* d = sF + tid * SF_STRIDE + FDIM;
#pragma unroll
                for (int dd = 0; dd < JD; ++dd) d[dd] = bfbits(src[dd]);
            }
        }
        // stage transposed W2[j]: 256 chunks of 16B
        {
            const unsigned short* g = wt2 + (size_t)j * WT2_JOINT_ELEMS;
            int n  = tid >> 4;
            int k8 = (tid & 15) * 8;
            copy_chunk16(g + tid * 8, sW2 + n * SW2_STRIDE + k8);
        }

        float b1v[8];
#pragma unroll
        for (int nt = 0; nt < 8; ++nt) b1v[nt] = b1[j * HID + nt * 16 + n15];
#pragma unroll
        for (int nt = 0; nt < 8; ++nt) acc[nt] = zero8();

        stage_w1_tile(j, 0, 0);
        // ---- layer 1: K = 544 (features 512 | parent_rot 6 | zero pad) ----
        for (int kt = 0; kt < NKT; ++kt) {
            async_barrier();                               // tile kt resident
            if (kt + 1 < NKT) stage_w1_tile(j, kt + 1, (kt + 1) & 1);
            const unsigned short* wb = sWt + (kt & 1) * (HID * SW_STRIDE);
            v16bf a = load_a(sF, wave * 16, SF_STRIDE, kt * 32, lane);
#pragma unroll
            for (int nt = 0; nt < 8; ++nt) {
                v16bf bf = load_b(wb, nt * 16, SW_STRIDE, 0, lane);
                acc[nt] = __builtin_amdgcn_wmma_f32_16x16x32_bf16(
                    false, a, false, bf, (short)0, acc[nt], false, false);
            }
        }

        // ---- bias + ReLU; H -> wave-local bf16 strip (no barrier needed) ----
#pragma unroll
        for (int nt = 0; nt < 8; ++nt) {
            int N = nt * 16 + n15;
#pragma unroll
            for (int v = 0; v < 8; ++v) {
                float h = fmaxf(acc[nt][v] + b1v[nt], 0.0f);
                sH[(wave * 16 + hi8 + v) * SH_STRIDE + N] = bfbits(h);
            }
        }

        // ---- layer 2: P(16x6) = H(16x128) @ W2 (N padded to 16) ----
        v8f accp = zero8();
#pragma unroll
        for (int kt2 = 0; kt2 < 4; ++kt2) {
            v16bf a  = load_a(sH, wave * 16, SH_STRIDE, kt2 * 32, lane);
            v16bf bf = load_b(sW2, 0, SW2_STRIDE, kt2 * 32, lane);
            accp = __builtin_amdgcn_wmma_f32_16x16x32_bf16(
                false, a, false, bf, (short)0, accp, false, false);
        }
        if (n15 < JD) {
            float bv = b2[j * JD + n15];
#pragma unroll
            for (int v = 0; v < 8; ++v) {
                int r = wave * 16 + hi8 + v;
                sP[r * SP_STRIDE + j * JD + n15] = accp[v] + bv;
            }
        }
    }

    // ---- coalesced final store: sP [128][144] -> out ----
    __syncthreads();
    float* dst = out + (size_t)row0 * (NJ * JD);
    for (int idx = tid * 4; idx < ROWS * NJ * JD; idx += THREADS * 4) {
        int r = idx / (NJ * JD);
        int c = idx % (NJ * JD);
        *(float4*)(dst + (size_t)r * (NJ * JD) + c) =
            *(const float4*)(sP + r * SP_STRIDE + c);
    }
}

extern "C" void kernel_launch(void* const* d_in, const int* in_sizes, int n_in,
                              void* d_out, int out_size, void* d_ws, size_t ws_size,
                              hipStream_t stream) {
    (void)n_in; (void)ws_size; (void)out_size;
    const float* features = (const float*)d_in[0];
    const float* W1 = (const float*)d_in[1];
    const float* b1 = (const float*)d_in[2];
    const float* W2 = (const float*)d_in[3];
    const float* b2 = (const float*)d_in[4];
    float* out = (float*)d_out;

    unsigned short* wt1 = (unsigned short*)d_ws;        // 3.34 MB
    unsigned short* wt2 = wt1 + WT1_ELEMS;              // + 96 KB

    // one-shot weight convert/transpose (stream-ordered before main kernel)
    hipLaunchKernelGGL(prep_w1_kernel, dim3(NJ * NKT), dim3(THREADS), 0, stream,
                       W1, wt1);
    hipLaunchKernelGGL(prep_w2_kernel, dim3(NJ), dim3(THREADS), 0, stream,
                       W2, wt2);

    const int B = in_sizes[0] / FDIM;   // 16384
    dim3 grid(B / ROWS);                // 128 workgroups
    hipLaunchKernelGGL(smpl_chain_kernel, grid, dim3(THREADS), LDS_BYTES, stream,
                       features, wt1, wt2, b1, b2, out);
}
